// CrossAttention_41497974014052
// MI455X (gfx1250) — matrix operand order repaired and checked by reading, not compiled, not run on hardware
//
#include <hip/hip_runtime.h>
#include <hip/hip_bf16.h>
#include <stdint.h>

typedef _Float16 v16h __attribute__((ext_vector_type(16)));
typedef _Float16 v8h  __attribute__((ext_vector_type(8)));
typedef float    v8f  __attribute__((ext_vector_type(8)));
typedef float    v4f  __attribute__((ext_vector_type(4)));
typedef uint32_t v4u  __attribute__((ext_vector_type(4)));
typedef uint32_t v8u  __attribute__((ext_vector_type(8)));

#define CDIM 256
#define HDIM 128
#define WDIM 128
#define BDIM 8
// HEAD_SIZE = 256/8 = 32 -> scale = 1/sqrt(32)
#define ATTN_SCALE 0.17677669529663687f

// LDS row strides (elements): rows 16B-aligned, consecutive rows 4 banks apart.
#define S264 264   // halves, 256-wide K-major tiles (528B = 33*16B)
#define S136 136   // halves, 128-wide tiles (272B = 17*16B)  == TDM pad layout
#define S260 260   // floats, 256-wide f32 score rows (1040B = 65*16B)

static __device__ __forceinline__ v16h cat16(v8h lo, v8h hi) {
    return __builtin_shufflevector(lo, hi, 0, 1, 2, 3, 4, 5, 6, 7,
                                   8, 9, 10, 11, 12, 13, 14, 15);
}

// ---------------------------------------------------------------------------
// Tensor Data Mover: async 2D tile copy global f16 -> LDS.
// Copies `rows` rows of 128 halves (row stride 128 halves in memory) into LDS
// at byte offset lds_off, inserting 4 DWORDs (8 halves) of padding after every
// 64 DWORDs (one row) -> LDS row stride = 136 halves = S136.
// D# per CDNA5 ISA ch.8: group0 {count=1, lds_addr, global_addr, type=2},
// group1 {data_size=1(2B), pad_enable, pad_interval=5(64dw), pad_amount=3(4dw),
//         tensor_dim0=128, tensor_dim1=rows, tile_dim0=128, tile_dim1=rows,
//         tensor_dim0_stride=128}.  2D -> groups 2/3 omitted (NULL).
// ---------------------------------------------------------------------------
static __device__ __forceinline__ void tdm_load_rows_f16(
    uint32_t lds_off, const _Float16* gptr, uint32_t rows)
{
    const uint64_t ga = (uint64_t)(uintptr_t)gptr;
    v4u g0;
    g0[0] = 1u;                                   // count=1, user mode
    g0[1] = lds_off;                              // lds_addr (bytes)
    g0[2] = (uint32_t)ga;                         // global_addr[31:0]
    g0[3] = (uint32_t)(ga >> 32) | 0x80000000u;   // global_addr[56:32] | type=2
    v8u g1;
    g1[0] = (1u << 16) | (1u << 20) | (5u << 22) | (3u << 25);
    g1[1] = (uint32_t)(WDIM) << 16;               // tensor_dim0[15:0]=128
    g1[2] = (rows & 0xFFFFu) << 16;               // dim0[31:16]=0 | tensor_dim1
    g1[3] = (uint32_t)(WDIM) << 16;               // dim1[31:16]=0 | tile_dim0=128
    g1[4] = rows & 0xFFFFu;                       // tile_dim1 | tile_dim2=0
    g1[5] = (uint32_t)WDIM;                       // tensor_dim0_stride = 128
    g1[6] = 0u;
    g1[7] = 0u;
    asm volatile("tensor_load_to_lds %0, %1" :: "s"(g0), "s"(g1) : "memory");
}

// ---------------------------------------------------------------------------
// Kernel A: fused depthwise 3x3 conv + pointwise 1x1 (256x256 matmul).
// One workgroup per (b, h).  Conv result stored K-major (yT[w][c], f16) so B
// fragments are two ds_load_b128.  Each wave owns 2 o-tiles; 8 weight
// A-fragments per o-tile hoisted to registers.  Output: [B, H, C, W] f16.
// ---------------------------------------------------------------------------
__global__ __launch_bounds__(256) void qkv_proj_kernel(
    const float* __restrict__ x,      // [B, C, H, W]
    const float* __restrict__ dw_w,   // [C, 1, 3, 3]
    const float* __restrict__ dw_b,   // [C]
    const float* __restrict__ pw_w,   // [O=C, C]
    const float* __restrict__ pw_b,   // [C]
    _Float16* __restrict__ out)       // [B, H, C, W] f16
{
    __shared__ __align__(16) _Float16 yT[WDIM][S264];   // [w][c], K(=c)-major

    const int bh = blockIdx.x;
    const int b = bh >> 7, h = bh & (HDIM - 1);
    const int tid = threadIdx.x;

    // ---- depthwise 3x3, SAME padding; one channel per thread ----
    {
        const int c = tid;
        const float* wv = dw_w + c * 9;
        const float w00 = wv[0], w01 = wv[1], w02 = wv[2];
        const float w10 = wv[3], w11 = wv[4], w12 = wv[5];
        const float w20 = wv[6], w21 = wv[7], w22 = wv[8];
        const float bias = dw_b[c];
        const float* base = x + (size_t)(b * CDIM + c) * HDIM * WDIM;
        const float* r0 = (h > 0)        ? base + (size_t)(h - 1) * WDIM : nullptr;
        const float* r1 = base + (size_t)h * WDIM;
        const float* r2 = (h < HDIM - 1) ? base + (size_t)(h + 1) * WDIM : nullptr;
        const v4f z4 = {0.f, 0.f, 0.f, 0.f};
        v4f a  = r0 ? *(const v4f*)(r0) : z4;
        v4f bb = *(const v4f*)(r1);
        v4f cc = r2 ? *(const v4f*)(r2) : z4;
        float ap = 0.f, bp = 0.f, cp = 0.f;     // x[w-1] (SAME padding)
        for (int w0 = 0; w0 < WDIM; w0 += 4) {
            const bool more = (w0 + 4 < WDIM);
            const v4f an = (more && r0) ? *(const v4f*)(r0 + w0 + 4) : z4;
            const v4f bn = more ? *(const v4f*)(r1 + w0 + 4) : z4;
            const v4f cn = (more && r2) ? *(const v4f*)(r2 + w0 + 4) : z4;
            const float aL[6] = {ap, a[0], a[1], a[2], a[3], an[0]};
            const float bL[6] = {bp, bb[0], bb[1], bb[2], bb[3], bn[0]};
            const float cL[6] = {cp, cc[0], cc[1], cc[2], cc[3], cn[0]};
            #pragma unroll
            for (int e = 0; e < 4; ++e) {
                const float s = bias
                    + aL[e] * w00 + aL[e + 1] * w01 + aL[e + 2] * w02
                    + bL[e] * w10 + bL[e + 1] * w11 + bL[e + 2] * w12
                    + cL[e] * w20 + cL[e + 1] * w21 + cL[e + 2] * w22;
                yT[w0 + e][c] = (_Float16)s;
            }
            ap = a[3]; bp = bb[3]; cp = cc[3];
            a = an; bb = bn; cc = cn;
        }
    }
    __syncthreads();

    // ---- pointwise: out[o][w] = sum_c pw_w[o][c] * yT[w][c] + pw_b[o] ----
    const int wave = tid >> 5, lane = tid & 31;
    const int m = lane & 15, g = lane >> 4;

    _Float16* obase = out + (size_t)(b * HDIM + h) * CDIM * WDIM;

    for (int half = 0; half < 2; ++half) {
        const int ob = (wave * 2 + half) * 16;

        v16h Af[8];
        const float* arow = pw_w + (size_t)(ob + m) * CDIM;
        #pragma unroll
        for (int cs8 = 0; cs8 < 8; ++cs8) {
            const float* p0 = arow + cs8 * 32 + g * 8;
            const float* p1 = p0 + 16;
            const v4f f0 = *(const v4f*)p0, f1 = *(const v4f*)(p0 + 4);
            const v4f f2 = *(const v4f*)p1, f3 = *(const v4f*)(p1 + 4);
            v16h A;
            #pragma unroll
            for (int e = 0; e < 4; ++e) {
                A[e]      = (_Float16)f0[e];
                A[4 + e]  = (_Float16)f1[e];
                A[8 + e]  = (_Float16)f2[e];
                A[12 + e] = (_Float16)f3[e];
            }
            Af[cs8] = A;
        }
        v8f bias;
        #pragma unroll
        for (int r = 0; r < 8; ++r) bias[r] = pw_b[ob + r + 8 * g];

        for (int wt = 0; wt < 8; ++wt) {
            const int wb = wt * 16;
            v8f acc = bias;
            #pragma unroll
            for (int cs8 = 0; cs8 < 8; ++cs8) {
                const _Float16* bp = &yT[wb + m][cs8 * 32 + g * 8];
                acc = __builtin_amdgcn_wmma_f32_16x16x32_f16(
                    false, Af[cs8], false,
                    cat16(*(const v8h*)bp, *(const v8h*)(bp + 16)),
                    (short)0, acc, false, false);
            }
            #pragma unroll
            for (int r = 0; r < 8; ++r)
                obase[(size_t)(ob + r + 8 * g) * WDIM + wb + m] = (_Float16)acc[r];
        }
    }
}

// ---------------------------------------------------------------------------
// Kernel B: attention for one (b, h, 64-channel query block).
// K and Q tiles arrive via the Tensor Data Mover (async DMA, hardware row
// padding -> S136 stride); V is transposed manually (b128 global loads).
// Scores f32 in LDS; WMMA for Q.K^T and P.V.  ~218 KB LDS total.
// ---------------------------------------------------------------------------
__global__ __launch_bounds__(256) void attn_kernel(
    const _Float16* __restrict__ qg,  // [B, H, C, W] f16
    const _Float16* __restrict__ kg,
    const _Float16* __restrict__ vg,
    float* __restrict__ outp)         // [B, C, H, W] f32
{
    __shared__ __align__(16) _Float16 kl[CDIM][S136];   // [d][w]  (TDM-filled)
    __shared__ __align__(16) _Float16 vT[WDIM][S264];   // [w][d]
    __shared__ __align__(16) _Float16 ql[64][S136];     // [c][w]  (TDM-filled)
    __shared__ __align__(16) float    sl[64][S260];     // scores / probs
    __shared__ float redm[64][4];
    __shared__ float reds[64][4];

    const int bh = blockIdx.x;
    const int b = bh >> 7, h = bh & (HDIM - 1);
    const int cb = blockIdx.y;
    const int tid = threadIdx.x;

    const size_t base = (size_t)(b * HDIM + h) * CDIM * WDIM;

    // ---- async DMA of K (256x128) and Q block (64x128) into LDS ----
    if (tid == 0) {
        tdm_load_rows_f16((uint32_t)(uintptr_t)&kl[0][0], kg + base, CDIM);
        tdm_load_rows_f16((uint32_t)(uintptr_t)&ql[0][0],
                          qg + base + (size_t)cb * 64 * WDIM, 64);
    }

    // ---- V: transpose into vT[w][d] with b128 global loads ----
    for (int idx = tid; idx < CDIM * WDIM / 8; idx += 256) {
        const int d = idx >> 4, w0 = (idx & 15) * 8;
        const v8h vv = *(const v8h*)&vg[base + (size_t)d * WDIM + w0];
        #pragma unroll
        for (int e = 0; e < 8; ++e) vT[w0 + e][d] = vv[e];
    }

    if (tid == 0) __builtin_amdgcn_s_wait_tensorcnt(0);
    __syncthreads();

    const int wave = tid >> 5, lane = tid & 31;
    const int n = lane & 15, g = lane >> 4;

    // ---- scores[c][d] = scale * sum_w q[c][w] * k[d][w] ----
    {
        const int ct = wave >> 1;
        const int dtbase = (wave & 1) * 8;
        v16h Aq[4];
        #pragma unroll
        for (int ks4 = 0; ks4 < 4; ++ks4) {
            const _Float16* p = &ql[ct * 16 + n][ks4 * 32 + g * 8];
            Aq[ks4] = cat16(*(const v8h*)p, *(const v8h*)(p + 16));
        }
        for (int dti = 0; dti < 8; ++dti) {
            const int dt = dtbase + dti;
            v8f acc;
            #pragma unroll
            for (int r = 0; r < 8; ++r) acc[r] = 0.0f;
            #pragma unroll
            for (int ks4 = 0; ks4 < 4; ++ks4) {
                const _Float16* bp = &kl[dt * 16 + n][ks4 * 32 + g * 8];
                acc = __builtin_amdgcn_wmma_f32_16x16x32_f16(
                    false, Aq[ks4], false,
                    cat16(*(const v8h*)bp, *(const v8h*)(bp + 16)),
                    (short)0, acc, false, false);
            }
            #pragma unroll
            for (int r = 0; r < 8; ++r)
                sl[ct * 16 + r + 8 * g][dt * 16 + n] = acc[r] * ATTN_SCALE;
        }
    }
    __syncthreads();

    // ---- softmax over d (256), 4 threads per row ----
    {
        const int row = tid >> 2, col0 = (tid & 3) * 64;
        float* srow = &sl[row][0];
        float mx = -3.4e38f;
        for (int j = 0; j < 64; ++j) mx = fmaxf(mx, srow[col0 + j]);
        redm[row][tid & 3] = mx;
        __syncthreads();
        const float rm = fmaxf(fmaxf(redm[row][0], redm[row][1]),
                               fmaxf(redm[row][2], redm[row][3]));
        float sum = 0.0f;
        for (int j = 0; j < 64; ++j) {
            const float e = __expf(srow[col0 + j] - rm);
            srow[col0 + j] = e;
            sum += e;
        }
        reds[row][tid & 3] = sum;
        __syncthreads();
        const float inv = 1.0f / (reds[row][0] + reds[row][1] +
                                  reds[row][2] + reds[row][3]);
        for (int j = 0; j < 64; ++j) srow[col0 + j] *= inv;
    }
    __syncthreads();

    // ---- out[c][w] = sum_d probs[c][d] * v[d][w] ----
    {
        const int ct = wave >> 1;
        const int wtbase = (wave & 1) * 4;
        v16h Ap[8];
        #pragma unroll
        for (int ds8 = 0; ds8 < 8; ++ds8) {
            const float* p0 = &sl[ct * 16 + n][ds8 * 32 + g * 8];
            const float* p1 = p0 + 16;
            const v4f f0 = *(const v4f*)p0, f1 = *(const v4f*)(p0 + 4);
            const v4f f2 = *(const v4f*)p1, f3 = *(const v4f*)(p1 + 4);
            v16h A;
            #pragma unroll
            for (int e = 0; e < 4; ++e) {
                A[e]      = (_Float16)f0[e];
                A[4 + e]  = (_Float16)f1[e];
                A[8 + e]  = (_Float16)f2[e];
                A[12 + e] = (_Float16)f3[e];
            }
            Ap[ds8] = A;
        }
        for (int wti = 0; wti < 4; ++wti) {
            const int wb = (wtbase + wti) * 16;
            v8f acc;
            #pragma unroll
            for (int r = 0; r < 8; ++r) acc[r] = 0.0f;
            #pragma unroll
            for (int ds8 = 0; ds8 < 8; ++ds8) {
                const _Float16* bp = &vT[wb + n][ds8 * 32 + g * 8];
                acc = __builtin_amdgcn_wmma_f32_16x16x32_f16(
                    false, Ap[ds8], false,
                    cat16(*(const v8h*)bp, *(const v8h*)(bp + 16)),
                    (short)0, acc, false, false);
            }
            #pragma unroll
            for (int r = 0; r < 8; ++r) {
                const int cglob = cb * 64 + ct * 16 + r + 8 * g;
                outp[((size_t)(b * CDIM + cglob) * HDIM + h) * WDIM + wb + n] = acc[r];
            }
        }
    }
}

// ---------------------------------------------------------------------------
extern "C" void kernel_launch(void* const* d_in, const int* in_sizes, int n_in,
                              void* d_out, int out_size, void* d_ws, size_t ws_size,
                              hipStream_t stream) {
    (void)in_sizes; (void)n_in; (void)out_size; (void)ws_size;
    const float* hidden = (const float*)d_in[0];
    const float* ctx    = (const float*)d_in[1];

    const size_t qkv_elems = (size_t)BDIM * HDIM * CDIM * WDIM;  // 64 MiB f16 each
    _Float16* q = (_Float16*)d_ws;
    _Float16* k = q + qkv_elems;
    _Float16* v = k + qkv_elems;   // 192 MiB of workspace total

    dim3 gridA(BDIM * HDIM);
    qkv_proj_kernel<<<gridA, 256, 0, stream>>>(
        hidden, (const float*)d_in[2], (const float*)d_in[3],
        (const float*)d_in[4], (const float*)d_in[5], q);
    qkv_proj_kernel<<<gridA, 256, 0, stream>>>(
        ctx, (const float*)d_in[6], (const float*)d_in[7],
        (const float*)d_in[8], (const float*)d_in[9], k);
    qkv_proj_kernel<<<gridA, 256, 0, stream>>>(
        ctx, (const float*)d_in[10], (const float*)d_in[11],
        (const float*)d_in[12], (const float*)d_in[13], v);

    dim3 gridB(BDIM * HDIM, CDIM / 64);
    attn_kernel<<<gridB, 256, 0, stream>>>(q, k, v, (float*)d_out);
}
